// NeuralCRF_38543036514759
// MI455X (gfx1250) — matrix operand order repaired and checked by reading, not compiled, not run on hardware
//
#include <hip/hip_runtime.h>

// ---------------------------------------------------------------------------
// NeuralCRF for MI455X (gfx1250): bf16 WMMA GEMMs, LDS-resident recurrences.
//   B=32, S=128, NUM_TAG=64, HID=256, TOPK=16, U=512, V=65, 3H=768
// ---------------------------------------------------------------------------

typedef __attribute__((ext_vector_type(16))) __bf16 v16bf;
typedef __attribute__((ext_vector_type(8)))  __bf16 v8bf;
typedef __attribute__((ext_vector_type(8)))  float  v8f;

#define NB   32
#define NS   128
#define NTAG 64
#define NH   256
#define NK   16
#define NU2  512   // U = emission feature dim
#define NH3  768   // 3*HID
#define NV   65    // NUM_TAG + 1 (SOS)

#define GAS __attribute__((address_space(1)))   // global
#define LAS __attribute__((address_space(3)))   // LDS

typedef const GAS v8bf* gv8bfp;
typedef const LAS v8bf* lv8bfp;

// --- address-space forcing helpers (generic LDS addr low 32 bits == ds offset)
static __device__ __forceinline__ const GAS __bf16* gp_bf(const __bf16* p) {
  return (const GAS __bf16*)(unsigned long long)p;
}
static __device__ __forceinline__ const LAS __bf16* lp_bf(const __bf16* p) {
  return (const LAS __bf16*)(unsigned)(unsigned long long)p;
}
static __device__ __forceinline__ float gldf(const float* p) {
  return *(const GAS float*)(unsigned long long)p;
}
static __device__ __forceinline__ float ldsf(const float* p) {
  return *(const LAS float*)(unsigned)(unsigned long long)p;
}
static __device__ __forceinline__ int ldsi(const int* p) {
  return *(const LAS int*)(unsigned)(unsigned long long)p;
}
static __device__ __forceinline__ void stsf(float* p, float v) {
  *(LAS float*)(unsigned)(unsigned long long)p = v;
}
static __device__ __forceinline__ void stsbf(__bf16* p, __bf16 v) {
  *(LAS __bf16*)(unsigned)(unsigned long long)p = v;
}

static __device__ __forceinline__ v8f wmma_bf(v16bf a, v16bf b, v8f c) {
  // D = A(16x32 bf16) x B(32x16 bf16) + C(16x16 f32)
  return __builtin_amdgcn_wmma_f32_16x16x32_bf16(false, a, false, b, (short)0, c,
                                                 false, false);
}

// A-fragment (16-bit 16x32): lane m=lane&15; lanes<16 hold K[0..8)+[16..24),
// lanes>=16 hold K[8..16)+[24..32). rowbase = &src[row(lane)][k0].
static __device__ __forceinline__ v16bf a_frag_g(const __bf16* rowbase, int lane) {
  const GAS __bf16* p = gp_bf(rowbase) + ((lane & 16) ? 8 : 0);
  v8bf lo = *(gv8bfp)p;
  v8bf hi = *(gv8bfp)(p + 16);
  return __builtin_shufflevector(lo, hi, 0,1,2,3,4,5,6,7,8,9,10,11,12,13,14,15);
}
static __device__ __forceinline__ v16bf a_frag_l(const __bf16* rowbase, int lane) {
  const LAS __bf16* p = lp_bf(rowbase) + ((lane & 16) ? 8 : 0);
  v8bf lo = *(lv8bfp)p;
  v8bf hi = *(lv8bfp)(p + 16);
  return __builtin_shufflevector(lo, hi, 0,1,2,3,4,5,6,7,8,9,10,11,12,13,14,15);
}
// B-fragment (16-bit 32x16): lane col=lane&15; lanes<16 hold K[0..8)+[16..24),
// lanes>=16 hold K[8..16)+[24..32). colbase = &srcT[col(lane)][k0] (contiguous in K).
static __device__ __forceinline__ v16bf b_frag_g(const __bf16* colbase, int lane) {
  const GAS __bf16* p = gp_bf(colbase) + ((lane & 16) ? 8 : 0);
  v8bf lo = *(gv8bfp)p;
  v8bf hi = *(gv8bfp)(p + 16);
  return __builtin_shufflevector(lo, hi, 0,1,2,3,4,5,6,7,8,9,10,11,12,13,14,15);
}

static __device__ __forceinline__ float sigmoidf_(float x) {
  return 1.f / (1.f + expf(-x));
}

// ---------------------------------------------------------------------------
// One GRU time-step for ROWS rows (ROWS in {16,32}), 8 waves.
// hf   : LDS f32  [ROWS][256]  (old h in, new h out)
// hb   : LDS bf16 [ROWS][256]  (old h in, new h out; WMMA A source)
// xs   : LDS int  [ROWS]       previous tag per row
// gi_tab[x][768] = W_ih[:,x] + b_ih  (one-hot matmul as gather)
// whh_bf[768][256] row j = W_hh row j  => B[k, j] = whh_bf[j][k]
// Entry: hb/xs coherent (barrier done by caller). Exit: hf/hb updated+synced.
// ---------------------------------------------------------------------------
template <int ROWS>
static __device__ __forceinline__ void gru_step_rows(
    float* hf, __bf16* hb, const int* xs,
    const float* __restrict__ gi_tab, const __bf16* __restrict__ whh_bf,
    const float* __restrict__ bhh, int wave, int lane) {
  constexpr int RT  = ROWS / 16;      // row tiles
  constexpr int NT  = RT * 16;        // gate-column-tile triplets (256/16 per RT)
  constexpr int PER = NT / 8;         // triplets per wave
  const v8f vzero = {0.f, 0.f, 0.f, 0.f, 0.f, 0.f, 0.f, 0.f};

  v8f ar[PER], az[PER], an_[PER];
#pragma unroll
  for (int p = 0; p < PER; ++p) { ar[p] = vzero; az[p] = vzero; an_[p] = vzero; }

  const int lm = lane & 15;
#pragma unroll 1
  for (int kk = 0; kk < 8; ++kk) {          // K = 256, 32 per WMMA
    int k0 = kk * 32;
    v16bf af[RT];
#pragma unroll
    for (int rt = 0; rt < RT; ++rt)
      af[rt] = a_frag_l(hb + (rt * 16 + lm) * NH + k0, lane);
    // issue all B loads for this K-step up front (clause + overlap)
    v16bf brf[PER], bzf[PER], bnf[PER];
#pragma unroll
    for (int p = 0; p < PER; ++p) {
      int ct = (wave * PER + p) & 15;
      const __bf16* w0 = whh_bf + (size_t)(ct * 16 + lm) * NH + k0;
      brf[p] = b_frag_g(w0, lane);
      bzf[p] = b_frag_g(w0 + 256 * NH, lane);   // z-gate rows 256..511
      bnf[p] = b_frag_g(w0 + 512 * NH, lane);   // n-gate rows 512..767
    }
#pragma unroll
    for (int p = 0; p < PER; ++p) {
      int rt = (wave * PER + p) >> 4;
      ar[p]  = wmma_bf(af[rt], brf[p], ar[p]);
      az[p]  = wmma_bf(af[rt], bzf[p], az[p]);
      an_[p] = wmma_bf(af[rt], bnf[p], an_[p]);
    }
  }
  __syncthreads();   // all waves done reading hb

  const int mh = (lane >> 4) * 8;
#pragma unroll
  for (int p = 0; p < PER; ++p) {
    int tI = wave * PER + p;
    int rt = tI >> 4, ct = tI & 15;
#pragma unroll
    for (int v = 0; v < 8; ++v) {
      int m = v + mh;
      int row = rt * 16 + m;
      int col = ct * 16 + lm;
      int x = ldsi(xs + row);
      const float* g = gi_tab + (size_t)x * NH3 + col;
      float gir = gldf(g), giz = gldf(g + 256), gin = gldf(g + 512);
      float ghr = ar[p][v]  + gldf(bhh + col);
      float ghz = az[p][v]  + gldf(bhh + 256 + col);
      float ghn = an_[p][v] + gldf(bhh + 512 + col);
      float r  = sigmoidf_(gir + ghr);
      float z  = sigmoidf_(giz + ghz);
      float nn = tanhf(gin + r * ghn);
      float ho = ldsf(hf + row * NH + col);
      float hn = (1.f - z) * nn + z * ho;
      stsf(hf + row * NH + col, hn);
      stsbf(hb + row * NH + col, (__bf16)hn);
    }
  }
  __syncthreads();
}

// ---------------------------------------------------------------------------
// Prep: bf16 weight copies, gi_table, bf16 emissions, gold input sequence.
// ---------------------------------------------------------------------------
__global__ __launch_bounds__(256) void prep_kernel(
    const float* __restrict__ W_ih, const float* __restrict__ W_hh,
    const float* __restrict__ b_ih, const float* __restrict__ Wj,
    const float* __restrict__ Wd, const float* __restrict__ emissions,
    const int* __restrict__ tags,
    __bf16* __restrict__ whh_bf, __bf16* __restrict__ wj_bf,
    __bf16* __restrict__ wd_bf, float* __restrict__ gi_tab,
    __bf16* __restrict__ em_bf, int* __restrict__ x_gold) {
  long i0 = (long)blockIdx.x * 256 + threadIdx.x;
  long st = (long)gridDim.x * 256;
  for (long i = i0; i < (long)NH3 * NH; i += st) whh_bf[i] = (__bf16)W_hh[i];
  for (long i = i0; i < (long)NH * NH3; i += st) wj_bf[i]  = (__bf16)Wj[i];
  for (long i = i0; i < (long)NTAG * NH; i += st) wd_bf[i] = (__bf16)Wd[i];
  for (long i = i0; i < (long)NV * NH3; i += st) {
    long x = i / NH3, r = i % NH3;
    gi_tab[i] = W_ih[r * NV + x] + b_ih[r];
  }
  for (long i = i0; i < (long)NB * NS * NU2; i += st) em_bf[i] = (__bf16)emissions[i];
  for (long i = i0; i < (long)NB * NS; i += st) {
    long b = i / NS, t = i % NS;
    x_gold[i] = (t == 0) ? NTAG : tags[b * NS + (t - 1)];
  }
}

// ---------------------------------------------------------------------------
// Beam-search decode: one workgroup per batch, entire 128-step scan in-kernel.
// Per step: GRU (WMMA) -> joint proj (WMMA) -> score (WMMA) -> top-16 -> reorder.
// ---------------------------------------------------------------------------
__global__ __launch_bounds__(256) void decode_kernel(
    const float* __restrict__ gi_tab, const __bf16* __restrict__ whh_bf,
    const float* __restrict__ bhh, const __bf16* __restrict__ em_bf,
    const __bf16* __restrict__ wj_bf, const float* __restrict__ bj,
    const __bf16* __restrict__ wd_bf, const float* __restrict__ bd,
    int* __restrict__ left_g, int* __restrict__ bps_g) {
  __shared__ __attribute__((aligned(16))) float  hf[NK * NH];
  __shared__ __attribute__((aligned(16))) __bf16 hb[NK * NH];
  __shared__ __attribute__((aligned(16))) float  projf[NK * NH];
  __shared__ __attribute__((aligned(16))) __bf16 projb[NK * NH];
  __shared__ float grow[NK * NTAG];
  __shared__ float redv[256];
  __shared__ int   redi[256];
  __shared__ int   xs[NK];
  __shared__ float prev[NK];
  __shared__ int   ncls[NK], nbeam[NK];
  __shared__ float nval[NK];

  const int tid = threadIdx.x, lane = tid & 31, wave = tid >> 5;
  const int lm = lane & 15, mh = (lane >> 4) * 8;
  const int b = blockIdx.x;

  for (int i = tid; i < NK * NH; i += 256) { hf[i] = 0.f; hb[i] = (__bf16)0.f; }
  if (tid < NK) { xs[tid] = NTAG; prev[tid] = (tid == 0) ? 0.f : -1e30f; }
  __syncthreads();

#pragma unroll 1
  for (int t = 0; t < NS; ++t) {
    // ---- GRU step over the 16 beams ----
    gru_step_rows<NK>(hf, hb, xs, gi_tab, whh_bf, bhh, wave, lane);

    // ---- joint stage 1: proj = tanh([em | h] @ Wj^T + bj), 16x256 ----
    const __bf16* emrow = em_bf + (size_t)(b * NS + t) * NU2;  // shared by beams
    const int ct0 = wave * 2, ct1 = ct0 + 1;
    const v8f vzero = {0.f, 0.f, 0.f, 0.f, 0.f, 0.f, 0.f, 0.f};
    v8f acc0 = vzero, acc1 = vzero;
#pragma unroll 1
    for (int kk = 0; kk < 16; ++kk) {       // K 0..511: emission part (global)
      int k0 = kk * 32;
      v16bf a  = a_frag_g(emrow + k0, lane);
      v16bf b0 = b_frag_g(wj_bf + (size_t)(ct0 * 16 + lm) * NH3 + k0, lane);
      v16bf b1 = b_frag_g(wj_bf + (size_t)(ct1 * 16 + lm) * NH3 + k0, lane);
      acc0 = wmma_bf(a, b0, acc0);
      acc1 = wmma_bf(a, b1, acc1);
    }
#pragma unroll 1
    for (int kk = 0; kk < 8; ++kk) {        // K 512..767: hidden part (LDS)
      int k0 = kk * 32;
      v16bf a  = a_frag_l(hb + lm * NH + k0, lane);
      v16bf b0 = b_frag_g(wj_bf + (size_t)(ct0 * 16 + lm) * NH3 + NU2 + k0, lane);
      v16bf b1 = b_frag_g(wj_bf + (size_t)(ct1 * 16 + lm) * NH3 + NU2 + k0, lane);
      acc0 = wmma_bf(a, b0, acc0);
      acc1 = wmma_bf(a, b1, acc1);
    }
#pragma unroll
    for (int v = 0; v < 8; ++v) {
      int m = v + mh;
      int c0 = ct0 * 16 + lm;
      float p0 = tanhf(acc0[v] + gldf(bj + c0));
      float p1 = tanhf(acc1[v] + gldf(bj + c0 + 16));
      stsf(projf + m * NH + c0, p0);       stsbf(projb + m * NH + c0, (__bf16)p0);
      stsf(projf + m * NH + c0 + 16, p1);  stsbf(projb + m * NH + c0 + 16, (__bf16)p1);
    }
    __syncthreads();

    // ---- joint stage 2: grow = prev + proj @ Wd^T + bd, 16x64 ----
    if (wave < 4) {
      v8f acc = vzero;
#pragma unroll
      for (int kk = 0; kk < 8; ++kk) {      // K = 256
        int k0 = kk * 32;
        v16bf a  = a_frag_l(projb + lm * NH + k0, lane);
        v16bf bb = b_frag_g(wd_bf + (size_t)(wave * 16 + lm) * NH + k0, lane);
        acc = wmma_bf(a, bb, acc);
      }
#pragma unroll
      for (int v = 0; v < 8; ++v) {
        int m = v + mh;
        int col = wave * 16 + lm;
        stsf(grow + m * NTAG + col, ldsf(prev + m) + acc[v] + gldf(bd + col));
      }
    }
    __syncthreads();

    // ---- top-16 over 1024 grow entries (ties -> lowest index) ----
#pragma unroll 1
    for (int it = 0; it < NK; ++it) {
      float bv = -3.4e38f; int bi = 0;
      for (int i = tid; i < NK * NTAG; i += 256) {   // per-thread ascending scan
        float v = ldsf(grow + i);
        if (v > bv) { bv = v; bi = i; }
      }
      redv[tid] = bv; redi[tid] = bi;
      __syncthreads();
      for (int s2 = 128; s2 > 0; s2 >>= 1) {
        if (tid < s2) {
          float ov = redv[tid + s2]; int oi = redi[tid + s2];
          if (ov > redv[tid] || (ov == redv[tid] && oi < redi[tid])) {
            redv[tid] = ov; redi[tid] = oi;
          }
        }
        __syncthreads();
      }
      if (tid == 0) {
        int idx = redi[0];
        nval[it]  = redv[0];
        ncls[it]  = idx % NTAG;
        nbeam[it] = idx / NTAG;
        stsf(grow + idx, -3.4e38f);
      }
      __syncthreads();
    }

    // ---- record backpointers, reorder h by beam, advance carry ----
    if (tid < NK) {
      left_g[((size_t)t * NB + b) * NK + tid] = ncls[tid];
      bps_g [((size_t)t * NB + b) * NK + tid] = nbeam[tid];
    }
    for (int i = tid; i < NK * NH; i += 256) { projf[i] = hf[i]; projb[i] = hb[i]; }
    __syncthreads();
    for (int i = tid; i < NK * NH; i += 256) {
      int r = i >> 8, c = i & 255;
      int src = nbeam[r];
      hf[i] = projf[src * NH + c];
      hb[i] = projb[src * NH + c];
    }
    if (tid < NK) { xs[tid] = ncls[tid]; prev[tid] = nval[tid]; }
    __syncthreads();
  }
}

// ---------------------------------------------------------------------------
// Backtrace (per (b,beam) thread) + beam input sequences + tags to d_out.
// ---------------------------------------------------------------------------
__global__ __launch_bounds__(256) void backtrace_kernel(
    const int* __restrict__ left_g, const int* __restrict__ bps_g,
    const float* __restrict__ mask, int* __restrict__ st,
    int* __restrict__ x_beam, float* __restrict__ out_tags) {
  int id = blockIdx.x * 256 + threadIdx.x;
  if (id >= NB * NK) return;
  int b = id >> 4, k = id & 15;
  float L = 0.f;
  for (int s = 0; s < NS; ++s) L += mask[b * NS + s];
  int len = (int)(L + 0.5f);
  int anc = k;
  int* row = st + (size_t)id * NS;
  for (int j = NS - 1; j >= 0; --j) {
    int jn = (j + 1 < NS) ? (j + 1) : (NS - 1);
    int cand = (j == len - 1) ? k : bps_g[((size_t)jn * NB + b) * NK + anc];
    int res = 0;
    if (j <= len - 1) { res = left_g[((size_t)j * NB + b) * NK + cand]; anc = cand; }
    row[j] = res;
  }
  for (int t = 0; t < NS; ++t) {
    x_beam[(size_t)id * NS + t] = (t == 0) ? NTAG : row[t - 1];
    out_tags[(size_t)id * NS + t] = (float)row[t];
  }
}

// ---------------------------------------------------------------------------
// Teacher-forced GRU scan + fused joint scoring: 16 rows per WG, whole sequence.
// scoresum[n] = sum_t ( tanh([em|h_t] @ Wj^T + bj) . Wd[tag(n,t)] + bd[tag] )
// em_div: 1 for gold (row n == batch), 16 for beams (batch = n/16).
// ---------------------------------------------------------------------------
__global__ __launch_bounds__(256) void seq_score_kernel(
    const int* __restrict__ xseq, const int* __restrict__ tagseq,
    const float* __restrict__ gi_tab, const __bf16* __restrict__ whh_bf,
    const float* __restrict__ bhh, const __bf16* __restrict__ em_bf,
    const __bf16* __restrict__ wj_bf, const float* __restrict__ bj,
    const float* __restrict__ Wd, const float* __restrict__ bd,
    float* __restrict__ scoresum, int em_div) {
  __shared__ __attribute__((aligned(16))) float  hf[NK * NH];
  __shared__ __attribute__((aligned(16))) __bf16 hb[NK * NH];
  __shared__ __attribute__((aligned(16))) float  projf[NK * NH];
  __shared__ int   xs[NK];
  __shared__ float rowsum[NK];
  __shared__ float red[256];

  const int tid = threadIdx.x, lane = tid & 31, wave = tid >> 5;
  const int lm = lane & 15, mh = (lane >> 4) * 8;
  const int n0 = blockIdx.x * NK;
  const v8f vzero = {0.f, 0.f, 0.f, 0.f, 0.f, 0.f, 0.f, 0.f};

  for (int i = tid; i < NK * NH; i += 256) { hf[i] = 0.f; hb[i] = (__bf16)0.f; }
  if (tid < NK) rowsum[tid] = 0.f;
  __syncthreads();

#pragma unroll 1
  for (int t = 0; t < NS; ++t) {
    if (tid < NK) xs[tid] = xseq[(size_t)(n0 + tid) * NS + t];
    __syncthreads();

    gru_step_rows<NK>(hf, hb, xs, gi_tab, whh_bf, bhh, wave, lane);

    // joint stage 1 (per-row emission rows, unlike decode)
    const int ct0 = wave * 2, ct1 = ct0 + 1;
    v8f acc0 = vzero, acc1 = vzero;
    const int nrow = n0 + lm;
    const __bf16* emrow = em_bf + ((size_t)(nrow / em_div) * NS + t) * NU2;
#pragma unroll 1
    for (int kk = 0; kk < 16; ++kk) {       // K 0..511: emission part (global)
      int k0 = kk * 32;
      v16bf a  = a_frag_g(emrow + k0, lane);
      v16bf b0 = b_frag_g(wj_bf + (size_t)(ct0 * 16 + lm) * NH3 + k0, lane);
      v16bf b1 = b_frag_g(wj_bf + (size_t)(ct1 * 16 + lm) * NH3 + k0, lane);
      acc0 = wmma_bf(a, b0, acc0);
      acc1 = wmma_bf(a, b1, acc1);
    }
#pragma unroll 1
    for (int kk = 0; kk < 8; ++kk) {        // K 512..767: hidden part (LDS)
      int k0 = kk * 32;
      v16bf a  = a_frag_l(hb + lm * NH + k0, lane);
      v16bf b0 = b_frag_g(wj_bf + (size_t)(ct0 * 16 + lm) * NH3 + NU2 + k0, lane);
      v16bf b1 = b_frag_g(wj_bf + (size_t)(ct1 * 16 + lm) * NH3 + NU2 + k0, lane);
      acc0 = wmma_bf(a, b0, acc0);
      acc1 = wmma_bf(a, b1, acc1);
    }
#pragma unroll
    for (int v = 0; v < 8; ++v) {
      int m = v + mh;
      int c0 = ct0 * 16 + lm;
      stsf(projf + m * NH + c0,      tanhf(acc0[v] + gldf(bj + c0)));
      stsf(projf + m * NH + c0 + 16, tanhf(acc1[v] + gldf(bj + c0 + 16)));
    }
    __syncthreads();

    // fused stage 2: only the tag'd column of proj @ Wd^T is needed
    int r = tid >> 4, l16 = tid & 15;
    int tg = tagseq[(size_t)(n0 + r) * NS + t];
    float part = 0.f;
    for (int c = l16; c < NH; c += 16)
      part += ldsf(projf + r * NH + c) * gldf(Wd + (size_t)tg * NH + c);
    red[tid] = part;
    __syncthreads();
    if (l16 == 0) {
      float s = 0.f;
      for (int j = 0; j < 16; ++j) s += red[r * 16 + j];
      rowsum[r] += s + gldf(bd + tg);
    }
    __syncthreads();
  }
  if (tid < NK) scoresum[n0 + tid] = rowsum[tid];
}

// ---------------------------------------------------------------------------
// Final scalar loss (matches reference's scalar-NLL construction exactly).
// ---------------------------------------------------------------------------
__global__ __launch_bounds__(256) void finalize_kernel(
    const float* __restrict__ gold_sum, const float* __restrict__ beam_sum,
    const float* __restrict__ mask, float* __restrict__ out) {
  __shared__ float red[256];
  __shared__ float sG, sS, totLse;
  int tid = threadIdx.x;

  float a = 0.f;
  for (int i = tid; i < NB * NK; i += 256) a += beam_sum[i];
  red[tid] = a; __syncthreads();
  for (int s = 128; s > 0; s >>= 1) { if (tid < s) red[tid] += red[tid + s]; __syncthreads(); }
  if (tid == 0) sS = red[0] / (float)(NB * NK * NS);
  __syncthreads();

  red[tid] = (tid < NB) ? gold_sum[tid] : 0.f; __syncthreads();
  for (int s = 128; s > 0; s >>= 1) { if (tid < s) red[tid] += red[tid + s]; __syncthreads(); }
  if (tid == 0) sG = red[0] / (float)(NB * NS);
  __syncthreads();

  float d = sG, sm = sS;
  float lse = 0.f, L = 0.f;
  if (tid < NB) {
    float Lb = 0.f;
    for (int t = 0; t < NS; ++t) Lb += mask[tid * NS + t];
    float v0 = d * Lb, v1 = sm * Lb;
    float m = fmaxf(v0, v1);
    lse = m + logf(expf(v0 - m) + (float)NK * expf(v1 - m));
    L = Lb;
  }
  red[tid] = lse; __syncthreads();
  for (int s = 128; s > 0; s >>= 1) { if (tid < s) red[tid] += red[tid + s]; __syncthreads(); }
  if (tid == 0) totLse = red[0];
  __syncthreads();
  red[tid] = L; __syncthreads();
  for (int s = 128; s > 0; s >>= 1) { if (tid < s) red[tid] += red[tid + s]; __syncthreads(); }
  if (tid == 0) out[0] = totLse / (float)NB - d * red[0] / (float)NB;
}

// ---------------------------------------------------------------------------
extern "C" void kernel_launch(void* const* d_in, const int* in_sizes, int n_in,
                              void* d_out, int out_size, void* d_ws, size_t ws_size,
                              hipStream_t stream) {
  const float* emissions = (const float*)d_in[0];
  const int*   tags      = (const int*)d_in[1];
  const float* mask      = (const float*)d_in[2];
  const float* W_ih      = (const float*)d_in[3];
  const float* W_hh      = (const float*)d_in[4];
  const float* b_ih      = (const float*)d_in[5];
  const float* b_hh      = (const float*)d_in[6];
  const float* Wj        = (const float*)d_in[7];
  const float* bj        = (const float*)d_in[8];
  const float* Wd        = (const float*)d_in[9];
  const float* bd        = (const float*)d_in[10];
  float* out = (float*)d_out;

  char* w = (char*)d_ws;
  size_t off = 0;
  auto alloc = [&](size_t bytes) -> void* {
    void* p = w + off;
    off = (off + bytes + 255) & ~(size_t)255;
    return p;
  };
  __bf16* whh_bf  = (__bf16*)alloc((size_t)NH3 * NH * 2);
  __bf16* wj_bf   = (__bf16*)alloc((size_t)NH * NH3 * 2);
  __bf16* wd_bf   = (__bf16*)alloc((size_t)NTAG * NH * 2);
  float*  gi_tab  = (float*) alloc((size_t)NV * NH3 * 4);
  __bf16* em_bf   = (__bf16*)alloc((size_t)NB * NS * NU2 * 2);
  int*    x_gold  = (int*)   alloc((size_t)NB * NS * 4);
  int*    left_g  = (int*)   alloc((size_t)NS * NB * NK * 4);
  int*    bps_g   = (int*)   alloc((size_t)NS * NB * NK * 4);
  int*    st      = (int*)   alloc((size_t)NB * NK * NS * 4);
  int*    x_beam  = (int*)   alloc((size_t)NB * NK * NS * 4);
  float*  gold_sum = (float*)alloc((size_t)NB * 4);
  float*  beam_sum = (float*)alloc((size_t)NB * NK * 4);

  // 1) weight conversion, gi table, bf16 emissions, gold input sequence
  prep_kernel<<<1024, 256, 0, stream>>>(W_ih, W_hh, b_ih, Wj, Wd, emissions, tags,
                                        whh_bf, wj_bf, wd_bf, gi_tab, em_bf, x_gold);
  // 2) full beam-search decode: one WG per batch, 128 steps in-kernel
  decode_kernel<<<NB, 256, 0, stream>>>(gi_tab, whh_bf, b_hh, em_bf, wj_bf, bj,
                                        wd_bf, bd, left_g, bps_g);
  // 3) backtrace -> search_tags (d_out[1..]) + beam input sequences
  backtrace_kernel<<<2, 256, 0, stream>>>(left_g, bps_g, mask, st, x_beam, out + 1);
  // 4) gold-path teacher-forced GRU + fused joint score sums (32 rows)
  seq_score_kernel<<<NB / NK, 256, 0, stream>>>(x_gold, tags, gi_tab, whh_bf, b_hh,
                                                em_bf, wj_bf, bj, Wd, bd,
                                                gold_sum, 1);
  // 5) beam-path rescoring GRU + fused joint score sums (512 rows)
  seq_score_kernel<<<NB, 256, 0, stream>>>(x_beam, st, gi_tab, whh_bf, b_hh,
                                           em_bf, wj_bf, bj, Wd, bd,
                                           beam_sum, NK);
  // 6) scalar loss -> d_out[0]
  finalize_kernel<<<1, 256, 0, stream>>>(gold_sum, beam_sum, mask, out);
}
